// GRAND_33320356282951
// MI455X (gfx1250) — compile-verified, matrix operand order checked by reading, not run on hardware
//
#include <hip/hip_runtime.h>

#define NNODES 50000
#define DFEAT 32
#define NEDGE 1600000
#define NCLS 40
#define NLAY 4
#define SCALEQK 0.17677669529663687f   // 1/sqrt(32)

typedef __attribute__((ext_vector_type(16))) __bf16 v16bf;
typedef __attribute__((ext_vector_type(8)))  float  v8f;

// ---------------------------------------------------------------------------
// WMMA helpers: bf16 split (hi+lo) GEMM fragments, f32 accumulate.
// A layout (16-bit 16x32): lane L holds row M=L&15; K groups {0-7,16-23} for
// lanes 0-15, {8-15,24-31} for lanes 16-31. B symmetric with lane = column N.
// C/D f32 16x16: VGPR r, lane L -> M = r + 8*(L>>4), N = L&15.
// ---------------------------------------------------------------------------
__device__ __forceinline__ int kmap(int i, int kb) {
  return kb + ((i < 4) ? (2 * i) : (8 + 2 * i));
}

__device__ __forceinline__ v8f wmma_bf16x3(v16bf ahi, v16bf alo, v16bf bhi,
                                           v16bf blo, v8f c) {
  c = __builtin_amdgcn_wmma_f32_16x16x32_bf16(false, alo, false, bhi, (short)0, c, false, false);
  c = __builtin_amdgcn_wmma_f32_16x16x32_bf16(false, ahi, false, blo, (short)0, c, false, false);
  c = __builtin_amdgcn_wmma_f32_16x16x32_bf16(false, ahi, false, bhi, (short)0, c, false, false);
  return c;
}

__device__ __forceinline__ void load_a(const float* __restrict__ xr, int kb,
                                       v16bf& ahi, v16bf& alo) {
#pragma unroll
  for (int i = 0; i < 8; ++i) {
    int k0 = kmap(i, kb);
    float f0 = xr[k0], f1 = xr[k0 + 1];
    __bf16 h0 = (__bf16)f0, h1 = (__bf16)f1;
    ahi[2 * i] = h0; ahi[2 * i + 1] = h1;
    alo[2 * i] = (__bf16)(f0 - (float)h0);
    alo[2 * i + 1] = (__bf16)(f1 - (float)h1);
  }
}

__device__ __forceinline__ void load_b(const float* __restrict__ W, int ldw,
                                       int col, int kb, bool valid,
                                       v16bf& bhi, v16bf& blo) {
#pragma unroll
  for (int i = 0; i < 8; ++i) {
    int k0 = kmap(i, kb);
    float f0 = valid ? W[k0 * ldw + col] : 0.f;
    float f1 = valid ? W[(k0 + 1) * ldw + col] : 0.f;
    __bf16 h0 = (__bf16)f0, h1 = (__bf16)f1;
    bhi[2 * i] = h0; bhi[2 * i + 1] = h1;
    blo[2 * i] = (__bf16)(f0 - (float)h0);
    blo[2 * i + 1] = (__bf16)(f1 - (float)h1);
  }
}

// Q = X*Wq + bq ; K = X*Wk + bk     (one 16-row tile per wave, 12 WMMAs)
__global__ void __launch_bounds__(256) qk_gemm_kernel(
    const float* __restrict__ X,
    const float* __restrict__ Wq, const float* __restrict__ bq,
    const float* __restrict__ Wk, const float* __restrict__ bk,
    float* __restrict__ Q, float* __restrict__ K, int ntiles) {
  int wave = threadIdx.x >> 5;
  int lane = threadIdx.x & 31;
  int tile = blockIdx.x * 8 + wave;
  if (tile >= ntiles) return;                 // wave-uniform guard (EXEC all-1s for WMMA)
  int row0 = tile * 16;
  int nlane = lane & 15;
  int kb = (lane >> 4) * 8;
  int mbase = row0 + ((lane >> 4) << 3);

  v16bf ahi, alo;
  load_a(X + (size_t)(row0 + nlane) * DFEAT, kb, ahi, alo);

#pragma unroll
  for (int t = 0; t < 2; ++t) {
    int col = t * 16 + nlane;
    v16bf bhi, blo;
    v8f c;
    // Q half
    load_b(Wq, DFEAT, col, kb, true, bhi, blo);
    float bv = bq[col];
#pragma unroll
    for (int r = 0; r < 8; ++r) c[r] = bv;
    c = wmma_bf16x3(ahi, alo, bhi, blo, c);
#pragma unroll
    for (int r = 0; r < 8; ++r) Q[(size_t)(mbase + r) * DFEAT + col] = c[r];
    // K half
    load_b(Wk, DFEAT, col, kb, true, bhi, blo);
    bv = bk[col];
#pragma unroll
    for (int r = 0; r < 8; ++r) c[r] = bv;
    c = wmma_bf16x3(ahi, alo, bhi, blo, c);
#pragma unroll
    for (int r = 0; r < 8; ++r) K[(size_t)(mbase + r) * DFEAT + col] = c[r];
  }
}

// out = X*Wr + br   ([N,32]x[32,40]; 3 col tiles, last one store-masked)
__global__ void __launch_bounds__(256) readout_kernel(
    const float* __restrict__ X, const float* __restrict__ Wr,
    const float* __restrict__ br, float* __restrict__ out, int ntiles) {
  int wave = threadIdx.x >> 5;
  int lane = threadIdx.x & 31;
  int tile = blockIdx.x * 8 + wave;
  if (tile >= ntiles) return;
  int row0 = tile * 16;
  int nlane = lane & 15;
  int kb = (lane >> 4) * 8;
  int mbase = row0 + ((lane >> 4) << 3);

  v16bf ahi, alo;
  load_a(X + (size_t)(row0 + nlane) * DFEAT, kb, ahi, alo);

#pragma unroll
  for (int t = 0; t < 3; ++t) {
    int col = t * 16 + nlane;
    bool valid = (col < NCLS);
    v16bf bhi, blo;
    load_b(Wr, NCLS, col, kb, valid, bhi, blo);
    v8f c;
    float bv = valid ? br[col] : 0.f;
#pragma unroll
    for (int r = 0; r < 8; ++r) c[r] = bv;
    c = wmma_bf16x3(ahi, alo, bhi, blo, c);   // whole wave executes WMMA
    if (valid) {                              // only stores are predicated
#pragma unroll
      for (int r = 0; r < 8; ++r) out[(size_t)(mbase + r) * NCLS + col] = c[r];
    }
  }
}

// ---------------------------------------------------------------------------
// Edge / softmax kernels (L2-resident, atomic-bound)
// ---------------------------------------------------------------------------
__global__ void set_x_kernel(const float* __restrict__ Xsrc, float* __restrict__ Xdst,
                             float* __restrict__ xall, int layer, int total) {
  int i = blockIdx.x * blockDim.x + threadIdx.x;
  if (i >= total) return;
  float v = Xsrc[i];
  if (Xdst) Xdst[i] = v;
  int node = i >> 5, f = i & 31;
  xall[(size_t)node * ((NLAY + 1) * DFEAT) + layer * DFEAT + f] = v;
}

__global__ void layer_init_kernel(float* __restrict__ m, float* __restrict__ s,
                                  float* __restrict__ Xnew, int n) {
  int i = blockIdx.x * blockDim.x + threadIdx.x;
  if (i < n) { m[i] = -__builtin_inff(); s[i] = 0.f; }
  if (i < n * DFEAT) Xnew[i] = 0.f;
}

// wave per edge: coalesced 128B row loads, wave32 shuffle reduce, float atomicMax
__global__ void __launch_bounds__(256) edge_logits_kernel(
    const int* __restrict__ src, const int* __restrict__ dst,
    const float* __restrict__ Q, const float* __restrict__ K,
    float* __restrict__ logits, float* __restrict__ m, int ne) {
  int e = (int)((blockIdx.x * blockDim.x + threadIdx.x) >> 5);
  if (e >= ne) return;
  int lane = threadIdx.x & 31;
  int sn = src[e], dn = dst[e];
  float v = Q[(size_t)dn * DFEAT + lane] * K[(size_t)sn * DFEAT + lane];
#pragma unroll
  for (int off = 16; off > 0; off >>= 1) v += __shfl_xor(v, off, 32);
  v *= SCALEQK;
  if (lane == 0) {
    logits[e] = v;
    if (v >= 0.f) atomicMax((int*)(m + dn), __float_as_int(v));
    else          atomicMin((unsigned int*)(m + dn), __float_as_uint(v));
  }
}

__global__ void edge_exp_kernel(const int* __restrict__ dst, float* __restrict__ logits,
                                const float* __restrict__ m, float* __restrict__ s, int ne) {
  int e = blockIdx.x * blockDim.x + threadIdx.x;
  if (e >= ne) return;
  int dn = dst[e];
  float v = __expf(logits[e] - m[dn]);
  logits[e] = v;
  atomicAdd(s + dn, v);
}

// wave per edge, lane = feature: Xnew[dst] += alpha * X[src]
__global__ void __launch_bounds__(256) edge_agg_kernel(
    const int* __restrict__ src, const int* __restrict__ dst,
    const float* __restrict__ logits, const float* __restrict__ s,
    const float* __restrict__ Xcur, float* __restrict__ Xnew, int ne) {
  int e = (int)((blockIdx.x * blockDim.x + threadIdx.x) >> 5);
  if (e >= ne) return;
  int lane = threadIdx.x & 31;
  int sn = src[e], dn = dst[e];
  float alpha = logits[e] / s[dn];
  atomicAdd(Xnew + (size_t)dn * DFEAT + lane, alpha * Xcur[(size_t)sn * DFEAT + lane]);
}

// ---------------------------------------------------------------------------
extern "C" void kernel_launch(void* const* d_in, const int* in_sizes, int n_in,
                              void* d_out, int out_size, void* d_ws, size_t ws_size,
                              hipStream_t stream) {
  const float* x  = (const float*)d_in[0];
  const int*   ei = (const int*)d_in[1];
  const float* Wq = (const float*)d_in[2];
  const float* bq = (const float*)d_in[3];
  const float* Wk = (const float*)d_in[4];
  const float* bk = (const float*)d_in[5];
  const float* Wr = (const float*)d_in[6];
  const float* br = (const float*)d_in[7];
  const int* src = ei;                // edge_index[0]
  const int* dst = ei + NEDGE;        // edge_index[1]

  float* out  = (float*)d_out;                       // [N, 40]
  float* xall = out + (size_t)NNODES * NCLS;         // [N, 5, 32]

  float* ws = (float*)d_ws;
  float* Q  = ws;
  float* K  = Q  + (size_t)NNODES * DFEAT;
  float* Xa = K  + (size_t)NNODES * DFEAT;
  float* Xb = Xa + (size_t)NNODES * DFEAT;
  float* lg = Xb + (size_t)NNODES * DFEAT;           // logits / e
  float* mg = lg + (size_t)NEDGE;                    // segment max
  float* sg = mg + (size_t)NNODES;                   // segment sum

  const int TPB = 256;
  const int nxthreads   = NNODES * DFEAT;            // 1.6M
  const int nxblocks    = (nxthreads + TPB - 1) / TPB;
  const int ntiles      = NNODES / 16;               // 3125 (exact)
  const int gemm_blocks = (ntiles + 7) / 8;          // 8 waves/block
  const int ewave_blocks = NEDGE / 8;                // wave per edge
  const int ethr_blocks  = (NEDGE + TPB - 1) / TPB;

  set_x_kernel<<<nxblocks, TPB, 0, stream>>>(x, Xa, xall, 0, nxthreads);

  float* Xcur = Xa;
  float* Xnew = Xb;
  for (int l = 0; l < NLAY; ++l) {
    qk_gemm_kernel<<<gemm_blocks, TPB, 0, stream>>>(Xcur, Wq, bq, Wk, bk, Q, K, ntiles);
    layer_init_kernel<<<nxblocks, TPB, 0, stream>>>(mg, sg, Xnew, NNODES);
    edge_logits_kernel<<<ewave_blocks, TPB, 0, stream>>>(src, dst, Q, K, lg, mg, NEDGE);
    edge_exp_kernel<<<ethr_blocks, TPB, 0, stream>>>(dst, lg, mg, sg, NEDGE);
    edge_agg_kernel<<<ewave_blocks, TPB, 0, stream>>>(src, dst, lg, sg, Xcur, Xnew, NEDGE);
    set_x_kernel<<<nxblocks, TPB, 0, stream>>>(Xnew, nullptr, xall, l + 1, nxthreads);
    float* t = Xcur; Xcur = Xnew; Xnew = t;          // DT=1 => X_new = agg
  }
  readout_kernel<<<gemm_blocks, TPB, 0, stream>>>(Xcur, Wr, br, out, ntiles);
}
